// DDSCTransformer_72267119722801
// MI455X (gfx1250) — matrix operand order repaired and checked by reading, not compiled, not run on hardware
//
#include <hip/hip_runtime.h>

// ---------------------------------------------------------------------------
// DDSC transformer for MI455X (gfx1250).
//  - f16 WMMA (16x16x32) for every GEMM (contraction dims are 32 / 512)
//  - weights pre-converted to f16 once per call (contiguous A-fragment loads)
//  - workgroup-cooperative flash attention: K/V tiles staged to LDS with
//    global_load_async_to_lds_b128 + s_wait_asynccnt (CDNA5 async path)
//  - D->B fragment repacking keeps score/hidden tiles in registers
//  - h intermediate stored position-major so WMMA outputs store as b128
// ---------------------------------------------------------------------------

typedef __attribute__((ext_vector_type(16))) _Float16 v16h;
typedef __attribute__((ext_vector_type(8)))  _Float16 v8h;
typedef __attribute__((ext_vector_type(8)))  float    v8f;

#define B_   32
#define CIN  6
#define C_   32
#define H_   8
#define L_   512
#define OC_  256   // C*H
#define DMC_ 128   // DM*C
#define EPS_ 1e-5f
#define SCALE_QK 0.42044820762685725f  // 32^-0.25

static __device__ __forceinline__ v8f wmma16(v16h a, v16h b, v8f c) {
  return __builtin_amdgcn_wmma_f32_16x16x32_f16(false, a, false, b, (short)0, c,
                                                false, false);
}

// A fragment (16x32 f16): lane holds row m=lane&15; K = base+j / base+8+j,
// base = (lane>>4)*16  (contiguous, matches B fragments)
static __device__ __forceinline__ v16h loadA_f16c(const _Float16* src,
                                                  int strideM) {
  int lane = threadIdx.x & 31;
  int m = lane & 15;
  int base = (lane >> 4) * 16;
  const _Float16* p = src + m * strideM + base;
  v8h lo = *(const v8h*)(p);
  v8h hi = *(const v8h*)(p + 8);
  v16h r;
#pragma unroll
  for (int j = 0; j < 8; ++j) { r[j] = lo[j]; r[j + 8] = hi[j]; }
  return r;
}

// B fragment (32x16 f16): lane holds col n=lane&15; K = kb+e, kb=(lane>>4)*16
static __device__ __forceinline__ v16h loadB_f32(const float* src, int strideN,
                                                 int strideK) {
  int lane = threadIdx.x & 31;
  int n = lane & 15;
  int kb = (lane >> 4) * 16;
  const float* p = src + n * strideN;
  v16h r;
#pragma unroll
  for (int e = 0; e < 16; ++e) r[e] = (_Float16)p[(kb + e) * strideK];
  return r;
}

static __device__ __forceinline__ v16h loadB_f16c(const _Float16* src,
                                                  int strideN) {
  int lane = threadIdx.x & 31;
  int n = lane & 15;
  int kb = (lane >> 4) * 16;
  return *(const v16h*)(src + n * strideN + kb);
}

// Repack two D fragments (rows 0..15 / 16..31 of a K=32 block, same columns)
// into a B fragment. shfl_xor(16) pairs lane n with lane n+16 (same column).
static __device__ __forceinline__ v16h packB_from_D(v8f d0, v8f d1) {
  int lane = threadIdx.x & 31;
  bool lo = lane < 16;
  v16h r;
#pragma unroll
  for (int i = 0; i < 8; ++i) {
    float o0 = __shfl_xor(d0[i], 16, 32);
    float o1 = __shfl_xor(d1[i], 16, 32);
    r[i]     = (_Float16)(lo ? d0[i] : o1);
    r[i + 8] = (_Float16)(lo ? o0 : d1[i]);
  }
  return r;
}

// ---------------------------------------------------------------------------

__global__ void k_wcvt(const float* __restrict__ src, _Float16* __restrict__ dst,
                       int n) {
  int i = blockIdx.x * blockDim.x + threadIdx.x;
  if (i < n) dst[i] = (_Float16)src[i];
}

__global__ void k_init(const float* __restrict__ x, const float* __restrict__ enc,
                       float* __restrict__ xA) {
  int idx = blockIdx.x * blockDim.x + threadIdx.x;  // over B*C*L
  if (idx >= B_ * C_ * L_) return;
  int l = idx % L_;
  int c = (idx / L_) % C_;
  int b = idx / (L_ * C_);
  const float* xb = x + b * CIN * L_ + l;
  const float* w = enc + c * CIN;
  float acc = 0.f;
#pragma unroll
  for (int i = 0; i < CIN; ++i) acc += w[i] * xb[i * L_];
  xA[idx] = acc;
}

// Pointwise conv 32 -> 256 (f16 weights): one wave per (b, 16-pos tile).
// Output stored position-major: h[b][l][oc]  -> contiguous b128 stores.
__global__ void k_pw(const float* __restrict__ xA,
                     const _Float16* __restrict__ wh,
                     _Float16* __restrict__ hbuf) {
  int q0 = (blockIdx.x & 31) * 16;
  int b = blockIdx.x >> 5;
  const float* xb = xA + b * C_ * L_ + q0;
  _Float16* hb = hbuf + ((size_t)b * L_ + q0) * OC_;
  v16h bx = loadB_f32(xb, 1, L_);  // B: cols = positions, K = channels
  int lane = threadIdx.x & 31;
  int n = lane & 15, rb = (lane >> 4) * 8;
#pragma unroll 4
  for (int ot = 0; ot < 16; ++ot) {
    v16h aw = loadA_f16c(wh + ot * 16 * C_, C_);  // A: rows = out channels
    v8f acc = {};
    acc = wmma16(aw, bx, acc);
    v8h st;
#pragma unroll
    for (int i = 0; i < 8; ++i) st[i] = (_Float16)acc[i];
    // lane writes channels ot*16+rb .. +7 at position q0+n (contiguous)
    *(v8h*)(hb + n * OC_ + ot * 16 + rb) = st;
  }
}

// Depthwise 3/15-tap convs + softmax gate; h is [b][l][oc] (channel-fastest
// thread mapping keeps every tap a fully coalesced 512B read).
__global__ void k_dw(const _Float16* __restrict__ hbuf,
                     const float* __restrict__ w3,
                     const float* __restrict__ w15,
                     const float* __restrict__ gate,
                     _Float16* __restrict__ dst, int transposed, float scale) {
  int idx = blockIdx.x * blockDim.x + threadIdx.x;  // over B*L*OC (oc fastest)
  if (idx >= B_ * L_ * OC_) return;
  int o = idx % OC_;
  int l = (idx / OC_) % L_;
  int b = idx / (OC_ * L_);
  float e0 = __expf(gate[0]), e1 = __expf(gate[1]);
  float g0 = e0 / (e0 + e1), g1 = 1.0f - g0;
  const _Float16* hb = hbuf + (size_t)b * L_ * OC_ + o;
  float a0 = 0.f, a1 = 0.f;
#pragma unroll
  for (int j = 0; j < 3; ++j) {
    int p = l + j - 1;
    float v = (p >= 0 && p < L_) ? (float)hb[p * OC_] : 0.f;
    a0 += v * w3[o * 3 + j];
  }
#pragma unroll
  for (int j = 0; j < 15; ++j) {
    int p = l + j - 7;
    float v = (p >= 0 && p < L_) ? (float)hb[p * OC_] : 0.f;
    a1 += v * w15[o * 15 + j];
  }
  float val = (g0 * a0 + g1 * a1) * scale;
  int hh = o >> 5, c = o & 31;
  if (transposed)  // [b,h][pos][c] for Q/K fragments (contiguous channel reads)
    dst[(((size_t)b * H_ + hh) * L_ + l) * C_ + c] = (_Float16)val;
  else             // [b,h][c][pos] for the V A-fragment
    dst[(((size_t)b * H_ + hh) * C_ + c) * L_ + l] = (_Float16)val;
}

// Workgroup-cooperative flash attention.
// Block = 128 threads (4 waves); each wave owns a 16-query tile (64 q / block).
// Per 32-key block: K and V tiles staged to LDS with async loads, consumed by
// all 4 waves as WMMA fragments.
__global__ void __launch_bounds__(128) k_attn(const _Float16* __restrict__ Qt,
                                              const _Float16* __restrict__ Kt,
                                              const _Float16* __restrict__ Vb,
                                              _Float16* __restrict__ Obuf) {
  __shared__ _Float16 ks[32 * 32];  // [kpos in block][c]
  __shared__ _Float16 vs[32 * 32];  // [c][kpos in block]
  int wave = threadIdx.x >> 5;
  int lane = threadIdx.x & 31;
  int tile = blockIdx.x & 7;           // 8 tiles of 64 queries
  int hh = (blockIdx.x >> 3) & 7;
  int b  = blockIdx.x >> 6;
  int q0 = tile * 64 + wave * 16;
  const _Float16* q_bh = Qt + (size_t)(b * H_ + hh) * L_ * C_;
  const _Float16* k_bh = Kt + (size_t)(b * H_ + hh) * L_ * C_;
  const _Float16* v_bh = Vb + (size_t)(b * H_ + hh) * C_ * L_;

  // staging assignment: thread copies one 16-byte chunk of K and of V
  int sr = threadIdx.x >> 2;           // 0..31 (kpos row for K, channel for V)
  int sc = (threadIdx.x & 3) * 8;      // 0,8,16,24 halves

  v16h bq = loadB_f16c(q_bh + q0 * C_, C_);  // cols = query pos, K = channels
  v8f o0 = {}, o1 = {};
  float mrun = -1e30f, ssum = 0.f;

  for (int k0 = 0; k0 < L_; k0 += 32) {
    // --- async global -> LDS staging of K[k0..k0+31][:] and V[:][k0..k0+31]
    {
      unsigned int lk = (unsigned int)(size_t)&ks[sr * 32 + sc];
      unsigned long long gk =
          (unsigned long long)(size_t)(k_bh + (k0 + sr) * C_ + sc);
      asm volatile("global_load_async_to_lds_b128 %0, %1, off"
                   :: "v"(lk), "v"(gk) : "memory");
      unsigned int lv = (unsigned int)(size_t)&vs[sr * 32 + sc];
      unsigned long long gv =
          (unsigned long long)(size_t)(v_bh + sr * L_ + k0 + sc);
      asm volatile("global_load_async_to_lds_b128 %0, %1, off"
                   :: "v"(lv), "v"(gv) : "memory");
      asm volatile("s_wait_asynccnt 0" ::: "memory");
    }
    __syncthreads();

    // S[kpos, qpos] = sum_c K[c,kpos] * Q[c,qpos]   (two 16-row tiles)
    v16h ak0 = loadA_f16c(ks, 32);
    v16h ak1 = loadA_f16c(ks + 16 * 32, 32);
    v8f s0 = {}, s1 = {};
    s0 = wmma16(ak0, bq, s0);
    s1 = wmma16(ak1, bq, s1);

    // online softmax over key positions (per query column = per lane pair)
    float lm = -1e30f;
#pragma unroll
    for (int i = 0; i < 8; ++i) { lm = fmaxf(lm, s0[i]); lm = fmaxf(lm, s1[i]); }
    lm = fmaxf(lm, __shfl_xor(lm, 16, 32));
    float mnew = fmaxf(mrun, lm);
    float corr = __expf(mrun - mnew);
    v8f p0, p1;
    float ps = 0.f;
#pragma unroll
    for (int i = 0; i < 8; ++i) {
      p0[i] = __expf(s0[i] - mnew);
      p1[i] = __expf(s1[i] - mnew);
      ps += p0[i] + p1[i];
    }
    ps += __shfl_xor(ps, 16, 32);
    ssum = ssum * corr + ps;
#pragma unroll
    for (int i = 0; i < 8; ++i) { o0[i] *= corr; o1[i] *= corr; }

    // O[c,q] += V[c, k0..k0+31] * P   (P repacked from score D fragments)
    v16h bp  = packB_from_D(p0, p1);
    v16h av0 = loadA_f16c(vs, 32);
    v16h av1 = loadA_f16c(vs + 16 * 32, 32);
    o0 = wmma16(av0, bp, o0);
    o1 = wmma16(av1, bp, o1);
    mrun = mnew;
    __syncthreads();  // all waves done reading before next staging overwrite
  }

  float inv = 1.0f / ssum;
  int n = lane & 15, rb = (lane >> 4) * 8;
  _Float16* ob = Obuf + ((size_t)b * L_ + q0 + n) * OC_ + hh * C_;
  v8h w0, w1;
#pragma unroll
  for (int i = 0; i < 8; ++i) {
    w0[i] = (_Float16)(o0[i] * inv);
    w1[i] = (_Float16)(o1[i] * inv);
  }
  *(v8h*)(ob + rb)      = w0;   // c = rb..rb+7
  *(v8h*)(ob + 16 + rb) = w1;   // c = 16+rb..16+rb+7
}

// unify (256 -> 32, f16 weights) + residual
__global__ void k_unify(const _Float16* __restrict__ Obuf,
                        const _Float16* __restrict__ wu,
                        const float* __restrict__ resid,
                        float* __restrict__ yres) {
  int q0 = (blockIdx.x & 31) * 16;
  int b = blockIdx.x >> 5;
  const _Float16* ob = Obuf + ((size_t)b * L_ + q0) * OC_;
  v8f acc0 = {}, acc1 = {};
  for (int s = 0; s < 8; ++s) {  // K = 256 channels in 8 steps
    v16h bo = loadB_f16c(ob + s * 32, OC_);
    v16h a0 = loadA_f16c(wu + s * 32, OC_);
    v16h a1 = loadA_f16c(wu + 16 * OC_ + s * 32, OC_);
    acc0 = wmma16(a0, bo, acc0);
    acc1 = wmma16(a1, bo, acc1);
  }
  int lane = threadIdx.x & 31;
  int n = lane & 15, rb = (lane >> 4) * 8;
  const float* rp = resid + b * C_ * L_ + q0 + n;
  float* yb = yres + b * C_ * L_ + q0 + n;
#pragma unroll
  for (int i = 0; i < 8; ++i) {
    int r0 = rb + i, r1 = 16 + rb + i;
    yb[r0 * L_] = acc0[i] + rp[r0 * L_];
    yb[r1 * L_] = acc1[i] + rp[r1 * L_];
  }
}

// InstanceNorm over L per (b,c): one wave per channel instance.
__global__ void k_inorm(const float* __restrict__ y, const float* __restrict__ g,
                        const float* __restrict__ bta, float* __restrict__ xout) {
  int wave = threadIdx.x >> 5;
  int lane = threadIdx.x & 31;
  int pair = blockIdx.x * 8 + wave;  // B*C = 1024 instances
  int b = pair >> 5, c = pair & 31;
  const float* row = y + ((size_t)b * C_ + c) * L_;
  float s = 0.f, s2 = 0.f;
#pragma unroll
  for (int j = 0; j < 16; ++j) {
    float v = row[lane + j * 32];
    s += v; s2 += v * v;
  }
#pragma unroll
  for (int off = 16; off > 0; off >>= 1) {
    s  += __shfl_xor(s, off, 32);
    s2 += __shfl_xor(s2, off, 32);
  }
  float mean = s * (1.0f / L_);
  float var  = s2 * (1.0f / L_) - mean * mean;
  float rs = rsqrtf(var + EPS_);
  float gg = g[c], bb = bta[c];
  float* orow = xout + ((size_t)b * C_ + c) * L_;
#pragma unroll
  for (int j = 0; j < 16; ++j) {
    float v = row[lane + j * 32];
    orow[lane + j * 32] = gg * (v - mean) * rs + bb;
  }
}

// FF: 32 -> 128 -> relu -> 32 (f16 weights), hidden stays in registers.
__global__ void k_ff(const float* __restrict__ xin,
                     const _Float16* __restrict__ w1,
                     const _Float16* __restrict__ w2,
                     float* __restrict__ yres) {
  int q0 = (blockIdx.x & 31) * 16;
  int b = blockIdx.x >> 5;
  const float* xb = xin + b * C_ * L_ + q0;
  v16h bx = loadB_f32(xb, 1, L_);
  v8f acc0 = {}, acc1 = {};
  for (int s = 0; s < 4; ++s) {  // dm = 128 in K-blocks of 32
    v8f h0 = {}, h1 = {};
    h0 = wmma16(loadA_f16c(w1 + (s * 32) * C_, C_), bx, h0);
    h1 = wmma16(loadA_f16c(w1 + (s * 32 + 16) * C_, C_), bx, h1);
#pragma unroll
    for (int i = 0; i < 8; ++i) {
      h0[i] = fmaxf(h0[i], 0.f);
      h1[i] = fmaxf(h1[i], 0.f);
    }
    v16h bh = packB_from_D(h0, h1);
    acc0 = wmma16(loadA_f16c(w2 + s * 32, DMC_), bh, acc0);
    acc1 = wmma16(loadA_f16c(w2 + 16 * DMC_ + s * 32, DMC_), bh, acc1);
  }
  int lane = threadIdx.x & 31;
  int n = lane & 15, rb = (lane >> 4) * 8;
  const float* rp = xb + n;
  float* yb = yres + b * C_ * L_ + q0 + n;
#pragma unroll
  for (int i = 0; i < 8; ++i) {
    int r0 = rb + i, r1 = 16 + rb + i;
    yb[r0 * L_] = acc0[i] + rp[r0 * L_];
    yb[r1 * L_] = acc1[i] + rp[r1 * L_];
  }
}

__global__ void k_out(const float* __restrict__ xA, const float* __restrict__ w,
                      float* __restrict__ out) {
  int idx = blockIdx.x * blockDim.x + threadIdx.x;  // over B*L
  if (idx >= B_ * L_) return;
  int l = idx % L_, b = idx / L_;
  const float* xb = xA + b * C_ * L_ + l;
  float acc = 0.f;
#pragma unroll
  for (int c = 0; c < C_; ++c) acc += w[c] * xb[c * L_];
  out[idx] = 1.0f / (1.0f + __expf(-acc));
}

// ---------------------------------------------------------------------------

extern "C" void kernel_launch(void* const* d_in, const int* in_sizes, int n_in,
                              void* d_out, int out_size, void* d_ws,
                              size_t ws_size, hipStream_t stream) {
  (void)in_sizes; (void)n_in; (void)out_size; (void)ws_size;

  // Inputs in setup_inputs() insertion order:
  // x, init_enc, then per block: (q,k,v each: pw,gate,dw0,dw1), unify,
  // n1g,n1b,n2g,n2b, ff1,ff2, then out.
  int idx = 0;
  const float* x_in = (const float*)d_in[idx++];
  const float* enc  = (const float*)d_in[idx++];
  struct DD { const float *pw, *gate, *dw0, *dw1; };
  struct Blk {
    DD q, k, v;
    const float *unify, *n1g, *n1b, *n2g, *n2b, *ff1, *ff2;
  } blk[6];
  for (int d = 0; d < 6; ++d) {
    DD* dds[3] = { &blk[d].q, &blk[d].k, &blk[d].v };
    for (int j = 0; j < 3; ++j) {
      dds[j]->pw   = (const float*)d_in[idx++];
      dds[j]->gate = (const float*)d_in[idx++];
      dds[j]->dw0  = (const float*)d_in[idx++];
      dds[j]->dw1  = (const float*)d_in[idx++];
    }
    blk[d].unify = (const float*)d_in[idx++];
    blk[d].n1g = (const float*)d_in[idx++];
    blk[d].n1b = (const float*)d_in[idx++];
    blk[d].n2g = (const float*)d_in[idx++];
    blk[d].n2b = (const float*)d_in[idx++];
    blk[d].ff1 = (const float*)d_in[idx++];
    blk[d].ff2 = (const float*)d_in[idx++];
  }
  const float* wout = (const float*)d_in[idx++];

  // Workspace carve-up (~47 MB; everything L2-resident on MI455X)
  char* ws = (char*)d_ws;
  float* xA = (float*)ws;            ws += (size_t)B_ * C_  * L_ * 4;
  _Float16* hbuf = (_Float16*)ws;    ws += (size_t)B_ * L_ * OC_ * 2;
  _Float16* Qt = (_Float16*)ws;      ws += (size_t)B_ * H_ * L_ * C_ * 2;
  _Float16* Kt = (_Float16*)ws;      ws += (size_t)B_ * H_ * L_ * C_ * 2;
  _Float16* Vb = (_Float16*)ws;      ws += (size_t)B_ * H_ * C_ * L_ * 2;
  _Float16* Obuf = (_Float16*)ws;    ws += (size_t)B_ * L_ * OC_ * 2;
  float* yres = (float*)ws;          ws += (size_t)B_ * C_ * L_ * 4;
  // f16 weight copies: per block pwq,pwk,pwv (8192 ea), unify 8192,
  // ff1 4096, ff2 4096
  _Float16* wh[6][6];
  for (int d = 0; d < 6; ++d) {
    const int sz[6] = {8192, 8192, 8192, 8192, 4096, 4096};
    for (int j = 0; j < 6; ++j) { wh[d][j] = (_Float16*)ws; ws += (size_t)sz[j] * 2; }
  }

  const int gPW  = B_ * (L_ / 16);          // 1024 waves
  const int gDW  = (B_ * OC_ * L_) / 256;   // 16384 blocks
  const int gATT = B_ * H_ * (L_ / 64);     // 2048 workgroups x 4 waves

  // one-time weight conversion to f16
  for (int d = 0; d < 6; ++d) {
    const float* srcs[6] = {blk[d].q.pw, blk[d].k.pw, blk[d].v.pw,
                            blk[d].unify, blk[d].ff1, blk[d].ff2};
    const int sz[6] = {8192, 8192, 8192, 8192, 4096, 4096};
    for (int j = 0; j < 6; ++j)
      k_wcvt<<<(sz[j] + 255) / 256, 256, 0, stream>>>(srcs[j], wh[d][j], sz[j]);
  }

  k_init<<<(B_ * C_ * L_ + 255) / 256, 256, 0, stream>>>(x_in, enc, xA);

  for (int d = 0; d < 6; ++d) {
    const Blk& bp = blk[d];
    k_pw<<<gPW, 32, 0, stream>>>(xA, wh[d][0], hbuf);
    k_dw<<<gDW, 256, 0, stream>>>(hbuf, bp.q.dw0, bp.q.dw1, bp.q.gate, Qt, 1, SCALE_QK);
    k_pw<<<gPW, 32, 0, stream>>>(xA, wh[d][1], hbuf);
    k_dw<<<gDW, 256, 0, stream>>>(hbuf, bp.k.dw0, bp.k.dw1, bp.k.gate, Kt, 1, SCALE_QK);
    k_pw<<<gPW, 32, 0, stream>>>(xA, wh[d][2], hbuf);
    k_dw<<<gDW, 256, 0, stream>>>(hbuf, bp.v.dw0, bp.v.dw1, bp.v.gate, Vb, 0, 1.0f);

    k_attn<<<gATT, 128, 0, stream>>>(Qt, Kt, Vb, Obuf);
    k_unify<<<gPW, 32, 0, stream>>>(Obuf, wh[d][3], xA, yres);
    k_inorm<<<(B_ * C_) / 8, 256, 0, stream>>>(yres, bp.n1g, bp.n1b, xA);
    k_ff<<<gPW, 32, 0, stream>>>(xA, wh[d][4], wh[d][5], yres);
    k_inorm<<<(B_ * C_) / 8, 256, 0, stream>>>(yres, bp.n2g, bp.n2b, xA);
  }

  k_out<<<(B_ * L_ + 255) / 256, 256, 0, stream>>>(xA, wout, (float*)d_out);
}
